// DAPTriAttStart_19112604467522
// MI455X (gfx1250) — compile-verified, matrix operand order checked by reading, not run on hardware
//
#include <hip/hip_runtime.h>
#include <hip/hip_bf16.h>

typedef __attribute__((ext_vector_type(16))) _Float16 v16h;
typedef __attribute__((ext_vector_type(8)))  _Float16 v8h;
typedef __attribute__((ext_vector_type(4)))  _Float16 v4h;
typedef __attribute__((ext_vector_type(8)))  float    v8f;

#define NN   256
#define DD   128
#define HH   4
#define CC   32
#define TOK  65536          // N*N tokens
#define TD   8388608ull     // TOK * DD

// Build a WMMA 16-half fragment from two contiguous 8-half (16B) chunks:
// elements [0..7] at p, elements [8..15] at p+16 halves.
__device__ __forceinline__ v16h load_frag(const _Float16* p) {
    v8h lo = *(const v8h*)p;
    v8h hi = *(const v8h*)(p + 16);
    v16h r;
#pragma unroll
    for (int e = 0; e < 8; ++e) { r[e] = lo[e]; r[8 + e] = hi[e]; }
    return r;
}

__device__ __forceinline__ v8f wmma32(v16h a, v16h b, v8f c) {
    return __builtin_amdgcn_wmma_f32_16x16x32_f16(false, a, false, b,
                                                  (short)0, c, false, false);
}

__device__ __forceinline__ float fast_rcp(float x) {
    return __builtin_amdgcn_rcpf(x);   // v_rcp_f32, avoids IEEE div expansion
}

// Async global->LDS copy of 16 bytes per lane (CDNA5 ASYNCcnt path).
// LDS offset = low 32 bits of the generic LDS pointer (ISA 10.2: flat LDS
// addresses map to LDS by truncating to [31:0]).
__device__ __forceinline__ void async_copy_b128(const _Float16* lds_dst,
                                                const _Float16* gsrc) {
    unsigned lds_off = (unsigned)(uintptr_t)lds_dst;
    asm volatile("global_load_async_to_lds_b128 %0, %1, off"
                 :: "v"(lds_off), "v"(gsrc) : "memory");
}
__device__ __forceinline__ void wait_async() {
    asm volatile("s_wait_asynccnt 0" ::: "memory");
}

// ---------------------------------------------------------------------------
// Kernel 0: transpose + f32->f16 convert the 5 GEMM weights into [N][K] order
// wt layout: [mat 0..4][n 0..127][k 0..127], mat: q,k,v,g,o
// ---------------------------------------------------------------------------
__global__ void k_wt(const float* __restrict__ wq, const float* __restrict__ wk,
                     const float* __restrict__ wv, const float* __restrict__ wg,
                     const float* __restrict__ wo, _Float16* __restrict__ wt) {
    int idx = blockIdx.x * 256 + threadIdx.x;   // 0 .. 81919
    int mat = idx >> 14;
    int r   = idx & 16383;
    int n   = r >> 7;
    int k   = r & 127;
    const float* W = (mat == 0) ? wq : (mat == 1) ? wk : (mat == 2) ? wv
                   : (mat == 3) ? wg : wo;
    wt[idx] = (_Float16)W[k * DD + n];
}

// ---------------------------------------------------------------------------
// Kernel 1: LayerNorm (wave per token) + triangle bias GEMV (D->H, f32)
// xn: f16 [TOK][128]; tri: f32 [H][256][256] indexed tri[h*65536 + token]
// ---------------------------------------------------------------------------
__global__ void k_ln_tri(const float* __restrict__ x, const float* __restrict__ ln_w,
                         const float* __restrict__ ln_b, const float* __restrict__ w_tri,
                         _Float16* __restrict__ xn, float* __restrict__ tri) {
    int wave = threadIdx.x >> 5;
    int lane = threadIdx.x & 31;
    int token = blockIdx.x * 8 + wave;

    const float* xp = x + (size_t)token * DD + lane * 4;
    float4 xv = *(const float4*)xp;

    float s = xv.x + xv.y + xv.z + xv.w;
#pragma unroll
    for (int m = 16; m >= 1; m >>= 1) s += __shfl_xor(s, m, 32);
    float mu = s * (1.0f / 128.0f);

    float d0 = xv.x - mu, d1 = xv.y - mu, d2 = xv.z - mu, d3 = xv.w - mu;
    float vs = d0 * d0 + d1 * d1 + d2 * d2 + d3 * d3;
#pragma unroll
    for (int m = 16; m >= 1; m >>= 1) vs += __shfl_xor(vs, m, 32);
    float rstd = rsqrtf(vs * (1.0f / 128.0f) + 1e-5f);

    float4 w = *(const float4*)(ln_w + lane * 4);
    float4 b = *(const float4*)(ln_b + lane * 4);
    float y[4];
    y[0] = d0 * rstd * w.x + b.x;
    y[1] = d1 * rstd * w.y + b.y;
    y[2] = d2 * rstd * w.z + b.z;
    y[3] = d3 * rstd * w.w + b.w;

    v4h yh;
#pragma unroll
    for (int t = 0; t < 4; ++t) yh[t] = (_Float16)y[t];
    *(v4h*)(xn + (size_t)token * DD + lane * 4) = yh;

    // tri[h] = sum_d xn[d] * w_tri[d][h],  w_tri laid out [128][4]
    float t0 = 0.f, t1 = 0.f, t2 = 0.f, t3 = 0.f;
#pragma unroll
    for (int t = 0; t < 4; ++t) {
        const float* wr = w_tri + (size_t)(lane * 4 + t) * HH;
        t0 += y[t] * wr[0]; t1 += y[t] * wr[1];
        t2 += y[t] * wr[2]; t3 += y[t] * wr[3];
    }
#pragma unroll
    for (int m = 16; m >= 1; m >>= 1) {
        t0 += __shfl_xor(t0, m, 32); t1 += __shfl_xor(t1, m, 32);
        t2 += __shfl_xor(t2, m, 32); t3 += __shfl_xor(t3, m, 32);
    }
    if (lane == 0) {
        tri[0 * TOK + token] = t0;
        tri[1 * TOK + token] = t1;
        tri[2 * TOK + token] = t2;
        tri[3 * TOK + token] = t3;
    }
}

// ---------------------------------------------------------------------------
// Kernel 2: projections  out[mat] = xn @ W[mat]  (mat: q,k,v,g via blockIdx.y)
// grid (4096, 4), block 256 (8 waves); wave w -> 16x16 tile at n0 = 16*w.
// A-tile staged to LDS with async global->LDS copies.
// ---------------------------------------------------------------------------
__global__ void k_proj(const _Float16* __restrict__ xn, const _Float16* __restrict__ wt,
                       const float* __restrict__ b_g, _Float16* __restrict__ outbase) {
    __shared__ _Float16 sA[16 * DD];

    int mat = blockIdx.y;
    const _Float16* W = wt + (size_t)mat * 16384;       // [n][k] f16
    _Float16* out = outbase + (size_t)mat * TD;

    int tid = threadIdx.x;
    int m0 = blockIdx.x * 16;
    async_copy_b128(sA + tid * 8, xn + (size_t)m0 * DD + tid * 8);
    wait_async();
    __syncthreads();

    int wave = tid >> 5, lane = tid & 31;
    int lm = lane & 15, lg = (lane >> 4) * 8;
    int n0 = wave * 16;

    v8f acc = {};
#pragma unroll
    for (int kb = 0; kb < DD; kb += 32) {
        v16h a = load_frag(sA + lm * DD + kb + lg);
        v16h b = load_frag(W + (size_t)(n0 + lm) * DD + kb + lg);
        acc = wmma32(a, b, acc);
    }

    int n = n0 + lm;
#pragma unroll
    for (int r = 0; r < 8; ++r) {
        int m = r + lg;
        float v = acc[r];
        if (mat == 0) v *= 0.17677669529663687f;                       // 1/sqrt(C)
        if (mat == 3) v = fast_rcp(1.0f + __expf(-(v + b_g[n])));      // sigmoid
        out[(size_t)(m0 + m) * DD + n] = (_Float16)v;
    }
}

// ---------------------------------------------------------------------------
// Kernel 3: attention for one (m-tile, head, row i).
// grid (16, 4, 256), block 256 (8 waves).
// Phase 1: scores (8 waves x 2 WMMA), phase 2: softmax, phase 3: P.V with all
// 8 waves (2 c-tiles x 4 K-partitions, partials reduced through LDS).
// ---------------------------------------------------------------------------
__global__ void k_attn(const _Float16* __restrict__ qbuf, const _Float16* __restrict__ kbuf,
                       const _Float16* __restrict__ vbuf, const _Float16* __restrict__ gbuf,
                       const float* __restrict__ tri, const float* __restrict__ mask,
                       _Float16* __restrict__ obuf) {
    __shared__ float    sS[16 * NN];   // scores f32; reused as P.V partials
    __shared__ _Float16 sP[16 * NN];   // softmax probs, f16
    __shared__ _Float16 sVt[CC * NN];  // V transposed [c][n], f16

    int mt = blockIdx.x, h = blockIdx.y, i = blockIdx.z;
    int tid = threadIdx.x, wave = tid >> 5, lane = tid & 31;
    int lm = lane & 15, lg = (lane >> 4) * 8;
    size_t rowbase = (size_t)i * NN;
    int m0 = mt * 16;

    // Stage V^T into LDS: sVt[c][n] = v[token=i*256+n][h*32+c]
    {
        int n = tid;
        const _Float16* vp = vbuf + (rowbase + n) * DD + h * CC;
#pragma unroll
        for (int c = 0; c < CC; c += 8) {
            v8h vv = *(const v8h*)(vp + c);
#pragma unroll
            for (int e = 0; e < 8; ++e) sVt[(c + e) * NN + n] = vv[e];
        }
    }

    // Scores: Q(16x32) . K^T, wave handles n-tiles 2w and 2w+1
    v16h aQ = load_frag(qbuf + (rowbase + m0 + lm) * DD + h * CC + lg);
#pragma unroll
    for (int t = 0; t < 2; ++t) {
        int n0 = (wave * 2 + t) * 16;
        v16h bK = load_frag(kbuf + (rowbase + n0 + lm) * DD + h * CC + lg);
        v8f s = {};
        s = wmma32(aQ, bK, s);
        int n = n0 + lm;
        float mb = 1e9f * (mask[rowbase + n] - 1.0f);
        const float* tp = tri + (size_t)h * TOK + (size_t)m0 * NN + n;
#pragma unroll
        for (int r = 0; r < 8; ++r) {
            int m = r + lg;
            sS[m * NN + n] = s[r] + tp[(size_t)m * NN] + mb;
        }
    }
    __syncthreads();

    // Softmax across 256 columns; wave handles rows 2w, 2w+1
#pragma unroll
    for (int rr = 0; rr < 2; ++rr) {
        int m = wave * 2 + rr;
        float vals[8];
        float mx = -3.0e38f;
#pragma unroll
        for (int j = 0; j < 8; ++j) {
            vals[j] = sS[m * NN + lane + 32 * j];
            mx = fmaxf(mx, vals[j]);
        }
#pragma unroll
        for (int s2 = 16; s2 >= 1; s2 >>= 1) mx = fmaxf(mx, __shfl_xor(mx, s2, 32));
        float sum = 0.f;
#pragma unroll
        for (int j = 0; j < 8; ++j) { vals[j] = __expf(vals[j] - mx); sum += vals[j]; }
#pragma unroll
        for (int s2 = 16; s2 >= 1; s2 >>= 1) sum += __shfl_xor(sum, s2, 32);
        float inv = fast_rcp(sum);
#pragma unroll
        for (int j = 0; j < 8; ++j)
            sP[m * NN + lane + 32 * j] = (_Float16)(vals[j] * inv);
    }
    __syncthreads();   // softmax reads of sS complete; sS reusable below

    // O = P . V  (16x256 @ 256x32): wave w -> c-tile (w&1), K-partition (w>>1)
    {
        int ct = wave & 1, kp = wave >> 1;
        v8f acc = {};
#pragma unroll
        for (int kk = 0; kk < 2; ++kk) {
            int kb = (kp * 2 + kk) * 32;
            v16h aP = load_frag(sP + lm * NN + kb + lg);
            v16h bV = load_frag(sVt + (ct * 16 + lm) * NN + kb + lg);
            acc = wmma32(aP, bV, acc);
        }
        // partials pf[ct][kp][m][n] in reused sS (2*4*16*16 floats)
        float* pf = sS;
#pragma unroll
        for (int r = 0; r < 8; ++r) {
            int m = r + lg;
            pf[((ct * 4 + kp) * 16 + m) * 16 + lm] = acc[r];
        }
    }
    __syncthreads();

    // Reduce 4 K-partials, gate, store (512 outputs, 2 per thread)
#pragma unroll
    for (int e0 = 0; e0 < 2; ++e0) {
        int e = tid + e0 * 256;
        int ct = e >> 8, m = (e >> 4) & 15, n = e & 15;
        const float* pf = sS;
        float s = pf[((ct * 4 + 0) * 16 + m) * 16 + n]
                + pf[((ct * 4 + 1) * 16 + m) * 16 + n]
                + pf[((ct * 4 + 2) * 16 + m) * 16 + n]
                + pf[((ct * 4 + 3) * 16 + m) * 16 + n];
        int c = ct * 16 + n;
        size_t tok = rowbase + m0 + m;
        float gv = (float)gbuf[tok * DD + h * CC + c];
        obuf[tok * DD + h * CC + c] = (_Float16)(gv * s);
    }
}

// ---------------------------------------------------------------------------
// Kernel 4: final projection  out = o @ w_o + b_o  (f32 output)
// ---------------------------------------------------------------------------
__global__ void k_out(const _Float16* __restrict__ obuf, const _Float16* __restrict__ wot,
                      const float* __restrict__ b_o, float* __restrict__ out) {
    __shared__ _Float16 sA[16 * DD];

    int tid = threadIdx.x;
    int m0 = blockIdx.x * 16;
    async_copy_b128(sA + tid * 8, obuf + (size_t)m0 * DD + tid * 8);
    wait_async();
    __syncthreads();

    int wave = tid >> 5, lane = tid & 31;
    int lm = lane & 15, lg = (lane >> 4) * 8;
    int n0 = wave * 16;

    v8f acc = {};
#pragma unroll
    for (int kb = 0; kb < DD; kb += 32) {
        v16h a = load_frag(sA + lm * DD + kb + lg);
        v16h b = load_frag(wot + (size_t)(n0 + lm) * DD + kb + lg);
        acc = wmma32(a, b, acc);
    }

    int n = n0 + lm;
    float bias = b_o[n];
#pragma unroll
    for (int r = 0; r < 8; ++r) {
        int m = r + lg;
        out[(size_t)(m0 + m) * DD + n] = acc[r] + bias;
    }
}

// ---------------------------------------------------------------------------
extern "C" void kernel_launch(void* const* d_in, const int* in_sizes, int n_in,
                              void* d_out, int out_size, void* d_ws, size_t ws_size,
                              hipStream_t stream) {
    const float* x     = (const float*)d_in[0];
    const float* mask  = (const float*)d_in[1];
    const float* ln_w  = (const float*)d_in[2];
    const float* ln_b  = (const float*)d_in[3];
    const float* w_tri = (const float*)d_in[4];
    const float* w_q   = (const float*)d_in[5];
    const float* w_k   = (const float*)d_in[6];
    const float* w_v   = (const float*)d_in[7];
    const float* w_g   = (const float*)d_in[8];
    const float* b_g   = (const float*)d_in[9];
    const float* w_o   = (const float*)d_in[10];
    const float* b_o   = (const float*)d_in[11];

    _Float16* xn   = (_Float16*)d_ws;
    _Float16* qbuf = xn + TD;
    _Float16* kbuf = qbuf + TD;
    _Float16* vbuf = kbuf + TD;
    _Float16* gbuf = vbuf + TD;
    _Float16* obuf = gbuf + TD;
    _Float16* wt   = obuf + TD;                // 5 * 16384 halves
    float*    tri  = (float*)(wt + 5 * 16384); // H * 65536 floats

    k_wt<<<320, 256, 0, stream>>>(w_q, w_k, w_v, w_g, w_o, wt);
    k_ln_tri<<<TOK / 8, 256, 0, stream>>>(x, ln_w, ln_b, w_tri, xn, tri);
    k_proj<<<dim3(TOK / 16, 4), 256, 0, stream>>>(xn, wt, b_g, qbuf);
    k_attn<<<dim3(16, HH, NN), 256, 0, stream>>>(qbuf, kbuf, vbuf, gbuf, tri, mask, obuf);
    k_out<<<TOK / 16, 256, 0, stream>>>(obuf, wt + 4 * 16384, b_o, (float*)d_out);
}
